// SpatialBertSelfAttention_64355789963933
// MI455X (gfx1250) — compile-verified, matrix operand order checked by reading, not run on hardware
//
#include <hip/hip_runtime.h>
#include <hip/hip_bf16.h>

// ---------------------------------------------------------------------------
// SpatialBertSelfAttention for MI455X (gfx1250, wave32, WMMA f16 16x16x32)
//
//   hidden[8,512,1024] f32, adj[8,512,512] i32, Wq/Wk/Wv[1024,1024], b*[1024]
//   out[8,512,1024] f32
//
// Pipeline (all f16 operands, f32 accumulation):
//   1) cvt_hidden  : hidden f32 -> h16 f16
//   2) cvt_weightT : W f32 -> wT f16 transposed [out][in] (B-fragment friendly)
//   3) qkv_gemm    : 2x4 register-blocked WMMA GEMM -> q16/k16 [b,h,n,64]
//                    (q pre-scaled by 1/8), vT16 [b,h,64,n]
//   4) attention   : flash-style online softmax, WMMA for QK^T and P.V
// ---------------------------------------------------------------------------

typedef __attribute__((ext_vector_type(16))) _Float16 v16h;
typedef __attribute__((ext_vector_type(8)))  _Float16 v8h;
typedef __attribute__((ext_vector_type(8)))  float    v8f;

#define NHEAD 16
#define NB    8
#define NSEQ  512
#define DMODEL 1024
#define HDIM  64

// relation->head allow mask (bit r = relation r allowed for this head),
// precomputed from the reference's per-head loop.
__device__ __constant__ unsigned short ALLOWED_MASK[NHEAD] = {
  0x3003, 0x3005, 0x3009, 0x3831, 0x3071, 0x30E1, 0x31C1, 0x3381,
  0x3701, 0x3E01, 0x3C11, 0x3831, 0x7001, 0x7001, 0x7001, 0x7001
};

// Load a 16-lane-half K-packed f16 fragment (A or B layout for 16x16x32):
// caller passes p = base + row*ld + k0 + (lane>>4)*8.  lo = K chunk 0..7(/8..15),
// hi = K chunk 16..23(/24..31) per the CDNA5 16-bit A/B VGPR layout.
__device__ inline v16h load_frag16(const _Float16* p) {
  v8h lo = *(const v8h*)(p);
  v8h hi = *(const v8h*)(p + 16);
  v16h r;
#pragma unroll
  for (int i = 0; i < 8; ++i) { r[i] = lo[i]; r[i + 8] = hi[i]; }
  return r;
}

__device__ inline v8f wmma_f16(v16h a, v16h b, v8f c) {
  return __builtin_amdgcn_wmma_f32_16x16x32_f16(
      /*neg_a=*/false, a, /*neg_b=*/false, b,
      /*c_mod=*/(short)0, c, /*reuse_a=*/false, /*reuse_b=*/false);
}

// ------------------------------- conversions -------------------------------

__global__ void cvt_hidden(const float* __restrict__ src,
                           _Float16* __restrict__ dst, int n) {
  int i = blockIdx.x * blockDim.x + threadIdx.x;
  if (i < n) dst[i] = (_Float16)src[i];
}

// wT[sel][n_out][k_in] = W_sel[k_in][n_out]   (transpose + f32->f16)
__global__ void cvt_weightT(const float* __restrict__ Wq,
                            const float* __restrict__ Wk,
                            const float* __restrict__ Wv,
                            _Float16* __restrict__ wT) {
  int i = blockIdx.x * blockDim.x + threadIdx.x;   // 0 .. 3*1024*1024-1
  int sel = i >> 20;
  int idx = i & ((1 << 20) - 1);
  int n = idx >> 10;
  int k = idx & (DMODEL - 1);
  const float* W = (sel == 0) ? Wq : ((sel == 1) ? Wk : Wv);
  wT[i] = (_Float16)W[(size_t)k * DMODEL + n];
}

// --------------------------------- QKV GEMM --------------------------------
// grid (128 row-blocks, 2 col-groups, 3 {q,k,v}), 256 threads = 8 waves.
// Each wave: 32x64 output block = 2x4 WMMA tiles.  Per K-step (32): loads
// 2 A frags + 4 B frags (6 KB/wave) for 8 WMMAs -> ~21 FLOP/byte from L2.

__global__ __launch_bounds__(256) void qkv_gemm(
    const _Float16* __restrict__ h16,   // [4096,1024]
    const _Float16* __restrict__ wT,    // [3][1024][1024]
    const float* __restrict__ bq, const float* __restrict__ bk,
    const float* __restrict__ bv,
    _Float16* __restrict__ q16,         // [b,h,n,64]  (pre-scaled 1/8)
    _Float16* __restrict__ k16,         // [b,h,n,64]
    _Float16* __restrict__ vT16)        // [b,h,64,n]
{
  const int lane = threadIdx.x & 31;
  const int wv   = threadIdx.x >> 5;
  const int half = lane >> 4;
  const int l16  = lane & 15;
  const int row0 = blockIdx.x * 32;                  // 0..4064
  const int col0 = (blockIdx.y * 8 + wv) * 64;       // 0..960
  const int sel  = blockIdx.z;

  const _Float16* w    = wT + (size_t)sel * DMODEL * DMODEL;
  const float*    bias = (sel == 0) ? bq : ((sel == 1) ? bk : bv);

  const _Float16* ap0 = h16 + (size_t)(row0 + l16) * DMODEL + half * 8;
  const _Float16* ap1 = ap0 + (size_t)16 * DMODEL;
  const _Float16* bp0 = w + (size_t)(col0 + l16) * DMODEL + half * 8;

  v8f acc[2][4];
#pragma unroll
  for (int m = 0; m < 2; ++m)
#pragma unroll
    for (int t = 0; t < 4; ++t) acc[m][t] = (v8f){};

#pragma unroll 2
  for (int kk = 0; kk < DMODEL; kk += 32) {
    v16h a0 = load_frag16(ap0 + kk);
    v16h a1 = load_frag16(ap1 + kk);
#pragma unroll
    for (int t = 0; t < 4; ++t) {
      v16h bf = load_frag16(bp0 + (size_t)(t * 16) * DMODEL + kk);
      acc[0][t] = wmma_f16(a0, bf, acc[0][t]);
      acc[1][t] = wmma_f16(a1, bf, acc[1][t]);
    }
  }

  const float qscale = (sel == 0) ? 0.125f : 1.0f;   // fold 1/sqrt(64) into Q
#pragma unroll
  for (int t = 0; t < 4; ++t) {
    const int col = col0 + t * 16 + l16;
    const float bcol = bias[col];
    const int hh = col >> 6, dd = col & (HDIM - 1);
#pragma unroll
    for (int m = 0; m < 2; ++m) {
#pragma unroll
      for (int r = 0; r < 8; ++r) {
        float val = (acc[m][t][r] + bcol) * qscale;
        int row = row0 + m * 16 + r + half * 8;      // C layout: M = r (+8)
        int bb = row >> 9, nn = row & (NSEQ - 1);
        if (sel == 2) {
          vT16[((size_t)(bb * NHEAD + hh) * HDIM + dd) * NSEQ + nn] =
              (_Float16)val;
        } else {
          _Float16* dst = (sel == 0) ? q16 : k16;
          dst[((size_t)(bb * NHEAD + hh) * NSEQ + nn) * HDIM + dd] =
              (_Float16)val;
        }
      }
    }
  }
}

// -------------------------------- attention --------------------------------
// One wave = 16 query rows of one (b,h); key loop in chunks of 32 with online
// softmax.  8 WMMAs per chunk (4 for S, 4 for P.V), 128 per wave.

__global__ __launch_bounds__(256) void attention(
    const _Float16* __restrict__ q16, const _Float16* __restrict__ k16,
    const _Float16* __restrict__ vT16, const int* __restrict__ adj,
    float* __restrict__ out)
{
  __shared__ _Float16 ldsP[8][16][32];   // per-wave P staging (C->A relayout)

  const int lane = threadIdx.x & 31;
  const int w    = threadIdx.x >> 5;
  const int half = lane >> 4;
  const int l16  = lane & 15;

  const int wid = blockIdx.x * 8 + w;    // 0..4095
  const int qt  = wid & 31;              // query tile 0..31
  const int h   = (wid >> 5) & (NHEAD - 1);
  const int b   = wid >> 9;

  const unsigned mask = ALLOWED_MASK[h];
  const _Float16* qb = q16 + ((size_t)(b * NHEAD + h) * NSEQ + qt * 16) * HDIM;
  const _Float16* kb = k16 + (size_t)(b * NHEAD + h) * NSEQ * HDIM;
  const _Float16* vb = vT16 + (size_t)(b * NHEAD + h) * HDIM * NSEQ;
  const int* adjb = adj + (size_t)b * NSEQ * NSEQ;

  // Q fragments (A layout), fixed across the key loop.
  v16h aQ0 = load_frag16(qb + (size_t)l16 * HDIM + half * 8);
  v16h aQ1 = load_frag16(qb + (size_t)l16 * HDIM + 32 + half * 8);

  v8f o0 = {}, o1 = {}, o2 = {}, o3 = {};
  float mrow[8], lrow[8];
#pragma unroll
  for (int r = 0; r < 8; ++r) { mrow[r] = -1e30f; lrow[r] = 0.0f; }

  for (int c = 0; c < NSEQ; c += 32) {
    // ---- S = Q.K^T for 16q x 32k (two 16x16 tiles) ----
    v8f s0 = {}, s1 = {};
    {
      v16h bk0 = load_frag16(kb + (size_t)(c + l16) * HDIM + half * 8);
      v16h bk1 = load_frag16(kb + (size_t)(c + l16) * HDIM + 32 + half * 8);
      s0 = wmma_f16(aQ0, bk0, s0);
      s0 = wmma_f16(aQ1, bk1, s0);
      v16h bk2 = load_frag16(kb + (size_t)(c + 16 + l16) * HDIM + half * 8);
      v16h bk3 = load_frag16(kb + (size_t)(c + 16 + l16) * HDIM + 32 + half * 8);
      s1 = wmma_f16(aQ0, bk2, s1);
      s1 = wmma_f16(aQ1, bk3, s1);
    }

    // ---- relation mask (scores += -1e9 where disallowed) ----
#pragma unroll
    for (int r = 0; r < 8; ++r) {
      int qrow = qt * 16 + r + half * 8;
      int rel0 = adjb[(size_t)qrow * NSEQ + c + l16];
      int rel1 = adjb[(size_t)qrow * NSEQ + c + 16 + l16];
      if (!((mask >> rel0) & 1u)) s0[r] += -1e9f;
      if (!((mask >> rel1) & 1u)) s1[r] += -1e9f;
    }

    // ---- online softmax update (row = (r, lane-half); cols across 16 lanes)
#pragma unroll
    for (int r = 0; r < 8; ++r) {
      float v = fmaxf(s0[r], s1[r]);
#pragma unroll
      for (int off = 8; off >= 1; off >>= 1) v = fmaxf(v, __shfl_xor(v, off, 32));
      float mnew = fmaxf(mrow[r], v);
      float sc = __expf(mrow[r] - mnew);
      float p0 = __expf(s0[r] - mnew);
      float p1 = __expf(s1[r] - mnew);
      float rs = p0 + p1;
#pragma unroll
      for (int off = 8; off >= 1; off >>= 1) rs += __shfl_xor(rs, off, 32);
      lrow[r] = lrow[r] * sc + rs;
      mrow[r] = mnew;
      o0[r] *= sc; o1[r] *= sc; o2[r] *= sc; o3[r] *= sc;
      s0[r] = p0;  s1[r] = p1;
    }

    // ---- relayout P through LDS: C layout -> A fragment ----
#pragma unroll
    for (int r = 0; r < 8; ++r) {
      ldsP[w][r + half * 8][l16]      = (_Float16)s0[r];
      ldsP[w][r + half * 8][16 + l16] = (_Float16)s1[r];
    }
    v16h aP = load_frag16(&ldsP[w][l16][half * 8]);

    // ---- O += P.V (4 hd tiles, B fragments from transposed V) ----
    v16h bv0 = load_frag16(vb + (size_t)(0  + l16) * NSEQ + c + half * 8);
    v16h bv1 = load_frag16(vb + (size_t)(16 + l16) * NSEQ + c + half * 8);
    v16h bv2 = load_frag16(vb + (size_t)(32 + l16) * NSEQ + c + half * 8);
    v16h bv3 = load_frag16(vb + (size_t)(48 + l16) * NSEQ + c + half * 8);
    o0 = wmma_f16(aP, bv0, o0);
    o1 = wmma_f16(aP, bv1, o1);
    o2 = wmma_f16(aP, bv2, o2);
    o3 = wmma_f16(aP, bv3, o3);
  }

  // ---- normalize and store fp32 output ----
#pragma unroll
  for (int r = 0; r < 8; ++r) {
    float inv = 1.0f / lrow[r];
    int qrow = qt * 16 + r + half * 8;
    float* op = out + ((size_t)(b * NSEQ + qrow) * DMODEL) + h * HDIM;
    op[0  + l16] = o0[r] * inv;
    op[16 + l16] = o1[r] * inv;
    op[32 + l16] = o2[r] * inv;
    op[48 + l16] = o3[r] * inv;
  }
}

// --------------------------------- launcher --------------------------------

extern "C" void kernel_launch(void* const* d_in, const int* in_sizes, int n_in,
                              void* d_out, int out_size, void* d_ws, size_t ws_size,
                              hipStream_t stream) {
  const float* hidden = (const float*)d_in[0];
  const int*   adj    = (const int*)d_in[1];
  const float* Wq     = (const float*)d_in[2];
  const float* bq     = (const float*)d_in[3];
  const float* Wk     = (const float*)d_in[4];
  const float* bk     = (const float*)d_in[5];
  const float* Wv     = (const float*)d_in[6];
  const float* bv     = (const float*)d_in[7];
  float* out = (float*)d_out;

  char* ws = (char*)d_ws;
  const size_t MB = 1024u * 1024u;
  _Float16* h16  = (_Float16*)(ws);             //  8 MB: [4096,1024]
  _Float16* wT   = (_Float16*)(ws + 8  * MB);   //  6 MB: [3,1024,1024]
  _Float16* q16  = (_Float16*)(ws + 14 * MB);   //  8 MB: [b,h,n,64]
  _Float16* k16  = (_Float16*)(ws + 22 * MB);   //  8 MB
  _Float16* vT16 = (_Float16*)(ws + 30 * MB);   //  8 MB: [b,h,64,n]

  const int nHid = NB * NSEQ * DMODEL;          // 4194304
  cvt_hidden<<<(nHid + 255) / 256, 256, 0, stream>>>(hidden, h16, nHid);
  cvt_weightT<<<(3 * DMODEL * DMODEL) / 256, 256, 0, stream>>>(Wq, Wk, Wv, wT);

  // 32-row blocks x 512-col blocks (8 waves x 64 cols) x {q,k,v}
  qkv_gemm<<<dim3(NB * NSEQ / 32, 2, 3), 256, 0, stream>>>(
      h16, wT, bq, bk, bv, q16, k16, vT16);

  // 8*16*32 = 4096 waves, 8 waves per block
  attention<<<512, 256, 0, stream>>>(q16, k16, vT16, adj, out);
}
